// TreeCRF_17549236372232
// MI455X (gfx1250) — matrix operand order repaired and checked by reading, not compiled, not run on hardware
//
#include <hip/hip_runtime.h>
#include <hip/hip_bf16.h>

// ---------------------------------------------------------------------------
// TreeCRF forward for MI455X (gfx1250, wave32, WMMA).
//   h  = relu(relu(X@W1+b1)@W2+b2)           [N,128]
//   out[:, :16]  = h@Wu+bu                   unary
//   out[:, 16:]  = concat(h[parent],h)@We+be edge
// bf16 WMMA; h staged as bf16 in d_ws (L2-resident, 102MB < 192MB L2).
// M=2 blocking (32 rows/wave) + WGP-cooperative double-buffered weight
// staging in LDS (global prefetch of nt+1 overlaps the nt WMMAs) + 4-deep
// batched B-fragment loads from LDS so ds waits amortize across 8 WMMAs.
// Streaming data (features in, logits out) uses non-temporal cache hints.
// ---------------------------------------------------------------------------

typedef __attribute__((ext_vector_type(16))) __bf16        v16bf;
typedef __attribute__((ext_vector_type(8)))  float         v8f;
typedef __attribute__((ext_vector_type(4)))  float         v4f;
typedef __attribute__((ext_vector_type(4)))  unsigned int  v4u;

union Frag {
    unsigned int u[8];   // 16 packed bf16
    v4u          q[2];   // two b128 transfers
    v16bf        v;      // WMMA operand view
};

// packed-weight layout inside d_ws (units: ushort/bf16 elements)
static constexpr int OFF_W1 = 0;        // 8 k-tiles * 8 n-tiles * 512
static constexpr int OFF_W2 = 32768;    // 4 * 8 * 512
static constexpr int OFF_WU = 49152;    // 4 * 1 * 512
static constexpr int OFF_WE = 51200;    // 8 * 16 * 512
static constexpr int OFF_H  = 131072;   // h buffer: N*128 bf16

__device__ __forceinline__ unsigned int bf_bits(float f) {
    unsigned int x = __float_as_uint(f);
    return (x + 0x7FFFu + ((x >> 16) & 1u)) >> 16;   // round-to-nearest-even
}
__device__ __forceinline__ unsigned int pack2(float lo, float hi) {
    return bf_bits(lo) | (bf_bits(hi) << 16);
}
__device__ __forceinline__ unsigned short bf1(float f) {
    return (unsigned short)bf_bits(f);
}
__device__ __forceinline__ void load_frag_lds(Frag& B, const unsigned short* wb,
                                              int kt, int lane) {
    const v4u* bp = (const v4u*)(wb + (kt << 9) + (lane << 4));
    B.q[0] = bp[0];
    B.q[1] = bp[1];
}

// ---------------------------------------------------------------------------
// Kernel 0: repack f32 row-major weights [K,N] into bf16 WMMA B-fragments.
//   lanes 0-15 : K = kt*32 + {0..7,16..23},  lanes 16-31 : K = kt*32 + {8..15,24..31}
//   VGPR j holds K = base+2j+(j>=4?8:0), +1 ; column N = nt*16 + (lane&15)
// ---------------------------------------------------------------------------
__global__ __launch_bounds__(256) void pack_weights_kernel(
    const float* __restrict__ W1, const float* __restrict__ W2,
    const float* __restrict__ Wu, const float* __restrict__ We,
    unsigned short* __restrict__ dst)
{
    const int t    = blockIdx.x * blockDim.x + threadIdx.x;
    const int frag = t >> 5;
    const int lane = t & 31;
    if (frag >= 228) return;   // 64 + 32 + 4 + 128 fragments

    const float* W;
    int Nn, KT, f;
    unsigned short* d;
    if (frag < 64)        { W = W1; Nn = 128; KT = 8; d = dst + OFF_W1; f = frag;       }
    else if (frag < 96)   { W = W2; Nn = 128; KT = 4; d = dst + OFF_W2; f = frag - 64;  }
    else if (frag < 100)  { W = Wu; Nn = 16;  KT = 4; d = dst + OFF_WU; f = frag - 96;  }
    else                  { W = We; Nn = 256; KT = 8; d = dst + OFF_WE; f = frag - 100; }

    const int kt    = f % KT;
    const int nt    = f / KT;
    const int n     = nt * 16 + (lane & 15);
    const int kbase = kt * 32 + (lane >> 4) * 8;
    unsigned short* o = d + f * 512 + lane * 16;
#pragma unroll
    for (int j = 0; j < 8; ++j) {
        float lo = W[(size_t)(kbase + 2 * j + (j >= 4 ? 8 : 0))     * Nn + n];
        float hi = W[(size_t)(kbase + 2 * j + (j >= 4 ? 8 : 0) + 1) * Nn + n];
        ((unsigned int*)o)[j] = pack2(lo, hi);
    }
}

// ---------------------------------------------------------------------------
// Kernel 1: MLP (2 layers) + unary head. 128 threads = 4 waves, one 32-node
// tile per wave (M=2). Weights staged per-nt in shared double-buffered LDS.
// ---------------------------------------------------------------------------
__global__ __launch_bounds__(128) void pass1_kernel(
    const float* __restrict__ X,  const float* __restrict__ b1,
    const float* __restrict__ b2, const float* __restrict__ bu,
    const unsigned short* __restrict__ wsu, float* __restrict__ out,
    unsigned short* __restrict__ hout, int nTiles32)
{
    __shared__ unsigned short lds_h[4][32 * 128];   // per-wave 8KB bf16 h tile
    __shared__ unsigned short wbuf[2][4096];        // 2 x 8KB weight staging

    const int  wave   = threadIdx.x >> 5;
    const int  lane   = threadIdx.x & 31;
    const int  tid    = threadIdx.x;
    const int  tile   = blockIdx.x * 4 + wave;
    const bool active = tile < nTiles32;
    const int  r0     = (active ? tile : 0) * 32;
    const int  lrow   = lane & 15;
    const int  hsel   = lane >> 4;
    unsigned short* hb = lds_h[wave];

    // ---- A fragments for GEMM1: features f32 -> bf16 (non-temporal reads) ----
    Frag A1[2][8];
#pragma unroll
    for (int m = 0; m < 2; ++m) {
        const float* xr = X + (size_t)(r0 + m * 16 + lrow) * 256 + hsel * 8;
#pragma unroll
        for (int kt = 0; kt < 8; ++kt) {
            v4f f0 = __builtin_nontemporal_load((const v4f*)(xr + kt * 32));
            v4f f1 = __builtin_nontemporal_load((const v4f*)(xr + kt * 32 + 4));
            v4f f2 = __builtin_nontemporal_load((const v4f*)(xr + kt * 32 + 16));
            v4f f3 = __builtin_nontemporal_load((const v4f*)(xr + kt * 32 + 20));
            A1[m][kt].u[0] = pack2(f0.x, f0.y);  A1[m][kt].u[1] = pack2(f0.z, f0.w);
            A1[m][kt].u[2] = pack2(f1.x, f1.y);  A1[m][kt].u[3] = pack2(f1.z, f1.w);
            A1[m][kt].u[4] = pack2(f2.x, f2.y);  A1[m][kt].u[5] = pack2(f2.z, f2.w);
            A1[m][kt].u[6] = pack2(f3.x, f3.y);  A1[m][kt].u[7] = pack2(f3.z, f3.w);
        }
    }

    // ================= GEMM1: [32x256]@[256x128] + b1, ReLU -> LDS ==========
    {
        const v4u* g0 = (const v4u*)(wsu + OFF_W1);      // nt=0 set: 512 v4u
#pragma unroll
        for (int i = 0; i < 4; ++i)
            ((v4u*)wbuf[0])[tid + (i << 7)] = g0[tid + (i << 7)];
    }
    __syncthreads();

    for (int nt = 0; nt < 8; ++nt) {
        const bool pf = (nt + 1) < 8;
        v4u t[4];
        const v4u* gn = (const v4u*)(wsu + OFF_W1 + ((nt + 1) << 12));
        if (pf) {
#pragma unroll
            for (int i = 0; i < 4; ++i) t[i] = gn[tid + (i << 7)];
        }
        if (active) {
            float bv = b1[nt * 16 + lrow];
            v8f a0 = {bv, bv, bv, bv, bv, bv, bv, bv};
            v8f a1 = a0;
            const unsigned short* wb = wbuf[nt & 1];
#pragma unroll
            for (int half = 0; half < 2; ++half) {
                Frag Bc[4];
#pragma unroll
                for (int j = 0; j < 4; ++j)
                    load_frag_lds(Bc[j], wb, half * 4 + j, lane);
#pragma unroll
                for (int j = 0; j < 4; ++j) {
                    const int kt = half * 4 + j;
                    a0 = __builtin_amdgcn_wmma_f32_16x16x32_bf16(
                        false, A1[0][kt].v, false, Bc[j].v, (short)0, a0, false, false);
                    a1 = __builtin_amdgcn_wmma_f32_16x16x32_bf16(
                        false, A1[1][kt].v, false, Bc[j].v, (short)0, a1, false, false);
                }
            }
            const int col = nt * 16 + lrow;
            const int mb  = hsel * 8;
#pragma unroll
            for (int g = 0; g < 8; ++g) {
                hb[(mb + g) * 128 + col]      = bf1(fmaxf(a0[g], 0.f));
                hb[(16 + mb + g) * 128 + col] = bf1(fmaxf(a1[g], 0.f));
            }
        }
        if (pf) {
            v4u* dn = (v4u*)wbuf[(nt + 1) & 1];
#pragma unroll
            for (int i = 0; i < 4; ++i) dn[tid + (i << 7)] = t[i];
        }
        __syncthreads();
    }

    // ================= GEMM2: [32x128]@[128x128] + b2, ReLU -> LDS ==========
    Frag A2[2][4];
    const unsigned short* hr0 = hb + lrow * 128 + hsel * 8;
    const unsigned short* hr1 = hb + (16 + lrow) * 128 + hsel * 8;
#pragma unroll
    for (int kt = 0; kt < 4; ++kt) {
        A2[0][kt].q[0] = *(const v4u*)(hr0 + kt * 32);
        A2[0][kt].q[1] = *(const v4u*)(hr0 + kt * 32 + 16);
        A2[1][kt].q[0] = *(const v4u*)(hr1 + kt * 32);
        A2[1][kt].q[1] = *(const v4u*)(hr1 + kt * 32 + 16);
    }
    {
        const v4u* g0 = (const v4u*)(wsu + OFF_W2);      // nt=0 set: 256 v4u
#pragma unroll
        for (int i = 0; i < 2; ++i)
            ((v4u*)wbuf[0])[tid + (i << 7)] = g0[tid + (i << 7)];
    }
    __syncthreads();

    for (int nt = 0; nt < 8; ++nt) {
        const bool pf = (nt + 1) < 8;
        v4u t[2];
        const v4u* gn = (const v4u*)(wsu + OFF_W2 + ((nt + 1) << 11));
        if (pf) {
#pragma unroll
            for (int i = 0; i < 2; ++i) t[i] = gn[tid + (i << 7)];
        }
        if (active) {
            float bv = b2[nt * 16 + lrow];
            v8f a0 = {bv, bv, bv, bv, bv, bv, bv, bv};
            v8f a1 = a0;
            const unsigned short* wb = wbuf[nt & 1];
            Frag Bc[4];
#pragma unroll
            for (int j = 0; j < 4; ++j) load_frag_lds(Bc[j], wb, j, lane);
#pragma unroll
            for (int kt = 0; kt < 4; ++kt) {
                a0 = __builtin_amdgcn_wmma_f32_16x16x32_bf16(
                    false, A2[0][kt].v, false, Bc[kt].v, (short)0, a0, false, false);
                a1 = __builtin_amdgcn_wmma_f32_16x16x32_bf16(
                    false, A2[1][kt].v, false, Bc[kt].v, (short)0, a1, false, false);
            }
            const int col = nt * 16 + lrow;
            const int mb  = hsel * 8;
#pragma unroll
            for (int g = 0; g < 8; ++g) {
                hb[(mb + g) * 128 + col]      = bf1(fmaxf(a0[g], 0.f));
                hb[(16 + mb + g) * 128 + col] = bf1(fmaxf(a1[g], 0.f));
            }
        }
        if (pf) {
            v4u* dn = (v4u*)wbuf[(nt + 1) & 1];
#pragma unroll
            for (int i = 0; i < 2; ++i) dn[tid + (i << 7)] = t[i];
        }
        __syncthreads();
    }

    // ---- coalesced copy of h (bf16) to global workspace for pass2 gather ----
    if (active) {
        const v4u* src = (const v4u*)hb;
        v4u* dst = (v4u*)(hout + (size_t)r0 * 128);
#pragma unroll
        for (int i = 0; i < 16; ++i) dst[i * 32 + lane] = src[i * 32 + lane];
    }

    // ================= unary head: [32x128]@[128x16] + bu ===================
    {
        const v4u* g0 = (const v4u*)(wsu + OFF_WU);      // 4 frags: 256 v4u
#pragma unroll
        for (int i = 0; i < 2; ++i)
            ((v4u*)wbuf[0])[tid + (i << 7)] = g0[tid + (i << 7)];
    }
    __syncthreads();

    if (active) {
        Frag A3[2][4];
#pragma unroll
        for (int kt = 0; kt < 4; ++kt) {
            A3[0][kt].q[0] = *(const v4u*)(hr0 + kt * 32);
            A3[0][kt].q[1] = *(const v4u*)(hr0 + kt * 32 + 16);
            A3[1][kt].q[0] = *(const v4u*)(hr1 + kt * 32);
            A3[1][kt].q[1] = *(const v4u*)(hr1 + kt * 32 + 16);
        }
        float bv = bu[lrow];
        v8f a0 = {bv, bv, bv, bv, bv, bv, bv, bv};
        v8f a1 = a0;
        const unsigned short* wb = wbuf[0];
        Frag Bc[4];
#pragma unroll
        for (int j = 0; j < 4; ++j) load_frag_lds(Bc[j], wb, j, lane);
#pragma unroll
        for (int kt = 0; kt < 4; ++kt) {
            a0 = __builtin_amdgcn_wmma_f32_16x16x32_bf16(
                false, A3[0][kt].v, false, Bc[kt].v, (short)0, a0, false, false);
            a1 = __builtin_amdgcn_wmma_f32_16x16x32_bf16(
                false, A3[1][kt].v, false, Bc[kt].v, (short)0, a1, false, false);
        }
        float* o0 = out + (size_t)(r0 + hsel * 8) * 272 + lrow;
        float* o1 = out + (size_t)(r0 + 16 + hsel * 8) * 272 + lrow;
#pragma unroll
        for (int g = 0; g < 8; ++g) {
            __builtin_nontemporal_store(a0[g], o0 + (size_t)g * 272);
            __builtin_nontemporal_store(a1[g], o1 + (size_t)g * 272);
        }
    }
}

// ---------------------------------------------------------------------------
// Kernel 2: edge head. A = concat(h[parent], h[self]) gathered from bf16 ws.
// 256 threads = 8 waves, one 32-node tile per wave (M=2). We fragments staged
// per-nt in shared double-buffered LDS (prefetch overlaps the 16 WMMAs).
// ---------------------------------------------------------------------------
__global__ __launch_bounds__(256) void pass2_kernel(
    const int* __restrict__ parent, const unsigned short* __restrict__ hsrc,
    const unsigned short* __restrict__ Wep, const float* __restrict__ be,
    float* __restrict__ out, int nTiles32)
{
    __shared__ unsigned short wbuf[2][4096];   // 2 x 8KB per-nt We fragments

    const int  wave   = threadIdx.x >> 5;
    const int  lane   = threadIdx.x & 31;
    const int  tid    = threadIdx.x;
    const int  tile   = blockIdx.x * 8 + wave;
    const bool active = tile < nTiles32;
    const int  r0     = (active ? tile : 0) * 32;
    const int  lrow   = lane & 15;
    const int  hsel   = lane >> 4;
    const int  n0     = r0 + lrow;
    const int  n1     = r0 + 16 + lrow;
    const int  p0     = parent[n0];
    const int  p1     = parent[n1];

    // A fragments: kt 0..3 from parent's h, kt 4..7 from own h (x2 row groups)
    Frag A[2][8];
    const unsigned short* hp0 = hsrc + (size_t)p0 * 128 + hsel * 8;
    const unsigned short* hs0 = hsrc + (size_t)n0 * 128 + hsel * 8;
    const unsigned short* hp1 = hsrc + (size_t)p1 * 128 + hsel * 8;
    const unsigned short* hs1 = hsrc + (size_t)n1 * 128 + hsel * 8;
#pragma unroll
    for (int kt = 0; kt < 4; ++kt) {
        A[0][kt].q[0]     = *(const v4u*)(hp0 + kt * 32);
        A[0][kt].q[1]     = *(const v4u*)(hp0 + kt * 32 + 16);
        A[0][4 + kt].q[0] = *(const v4u*)(hs0 + kt * 32);
        A[0][4 + kt].q[1] = *(const v4u*)(hs0 + kt * 32 + 16);
        A[1][kt].q[0]     = *(const v4u*)(hp1 + kt * 32);
        A[1][kt].q[1]     = *(const v4u*)(hp1 + kt * 32 + 16);
        A[1][4 + kt].q[0] = *(const v4u*)(hs1 + kt * 32);
        A[1][4 + kt].q[1] = *(const v4u*)(hs1 + kt * 32 + 16);
    }

    // prologue: stage nt=0 fragment set (8KB = 512 v4u, 256 threads x 2)
    {
        const v4u* g0 = (const v4u*)Wep;
        v4u a = g0[tid];
        v4u b = g0[tid + 256];
        ((v4u*)wbuf[0])[tid]       = a;
        ((v4u*)wbuf[0])[tid + 256] = b;
    }
    __syncthreads();

    // edge GEMM: [32x256]@[256x256] + be -> out[:, 16:272]
    for (int nt = 0; nt < 16; ++nt) {
        const bool pf = (nt + 1) < 16;
        v4u t0, t1;
        const v4u* gn = (const v4u*)(Wep + ((nt + 1) << 12));
        if (pf) { t0 = gn[tid]; t1 = gn[tid + 256]; }   // issue early (overlap)

        if (active) {
            float bv = be[nt * 16 + lrow];
            v8f a0 = {bv, bv, bv, bv, bv, bv, bv, bv};
            v8f a1 = a0;
            const unsigned short* wb = wbuf[nt & 1];
#pragma unroll
            for (int half = 0; half < 2; ++half) {
                Frag Bc[4];
#pragma unroll
                for (int j = 0; j < 4; ++j)
                    load_frag_lds(Bc[j], wb, half * 4 + j, lane);
#pragma unroll
                for (int j = 0; j < 4; ++j) {
                    const int kt = half * 4 + j;
                    a0 = __builtin_amdgcn_wmma_f32_16x16x32_bf16(
                        false, A[0][kt].v, false, Bc[j].v, (short)0, a0, false, false);
                    a1 = __builtin_amdgcn_wmma_f32_16x16x32_bf16(
                        false, A[1][kt].v, false, Bc[j].v, (short)0, a1, false, false);
                }
            }
            float* o0 = out + (size_t)(r0 + hsel * 8) * 272 + 16 + (nt << 4) + lrow;
            float* o1 = out + (size_t)(r0 + 16 + hsel * 8) * 272 + 16 + (nt << 4) + lrow;
#pragma unroll
            for (int g = 0; g < 8; ++g) {
                __builtin_nontemporal_store(a0[g], o0 + (size_t)g * 272);
                __builtin_nontemporal_store(a1[g], o1 + (size_t)g * 272);
            }
        }
        if (pf) {
            v4u* dn = (v4u*)wbuf[(nt + 1) & 1];
            dn[tid]       = t0;
            dn[tid + 256] = t1;
        }
        __syncthreads();
    }
}

// ---------------------------------------------------------------------------
extern "C" void kernel_launch(void* const* d_in, const int* in_sizes, int n_in,
                              void* d_out, int out_size, void* d_ws, size_t ws_size,
                              hipStream_t stream)
{
    const float* X      = (const float*)d_in[0];
    const int*   parent = (const int*)  d_in[1];
    const float* W1     = (const float*)d_in[2];
    const float* b1     = (const float*)d_in[3];
    const float* W2     = (const float*)d_in[4];
    const float* b2     = (const float*)d_in[5];
    const float* Wu     = (const float*)d_in[6];
    const float* bu     = (const float*)d_in[7];
    const float* We     = (const float*)d_in[8];
    const float* be     = (const float*)d_in[9];

    float* out = (float*)d_out;
    unsigned short* wsu  = (unsigned short*)d_ws;
    unsigned short* hbuf = wsu + OFF_H;

    const int N        = in_sizes[0] / 256;  // 400000 (divisible by 32)
    const int nTiles32 = N / 32;             // 12500

    pack_weights_kernel<<<(228 * 32 + 255) / 256, 256, 0, stream>>>(W1, W2, Wu, We, wsu);
    pass1_kernel<<<(nTiles32 + 3) / 4, 128, 0, stream>>>(X, b1, b2, bu, wsu, out, hbuf, nTiles32);
    pass2_kernel<<<(nTiles32 + 7) / 8, 256, 0, stream>>>(parent, hbuf, wsu + OFF_WE, be, out, nTiles32);
}
